// PWAC_66632122630633
// MI455X (gfx1250) — matrix operand order repaired and checked by reading, not compiled, not run on hardware
//
#include <hip/hip_runtime.h>

#define B_    16
#define C_    64
#define H_    64
#define W_    64
#define KC    16
#define OUT_  64
#define RANK_ 8
#define MLP_  32
#define F_    576
#define HW_   4096
#define ASTR  592   // LDS row stride (elements) for A tile, 16B-aligned, avoids conflicts

typedef __attribute__((ext_vector_type(16))) __bf16 v16bf;
typedef __attribute__((ext_vector_type(8)))  float  v8f;

__device__ __forceinline__ unsigned short f2bf(float f) {
  unsigned int u = __float_as_uint(f);
  unsigned int r = u + 0x7FFFu + ((u >> 16) & 1u);   // round-to-nearest-even
  return (unsigned short)(r >> 16);
}

// ---------------- stage 0: zero cluster counts ----------------
__global__ void k_zero(int* __restrict__ counts) {
  int t = blockIdx.x * blockDim.x + threadIdx.x;
  if (t < B_ * KC) counts[t] = 0;
}

// ---------------- stage 1: histogram labels ----------------
__global__ void k_count(const int* __restrict__ labels, int* __restrict__ counts) {
  int g = blockIdx.x * blockDim.x + threadIdx.x;   // 0 .. B*HW-1
  int b = g >> 12;
  int lab = labels[g];
  atomicAdd(&counts[b * KC + lab], 1);
}

// ---------------- stage 2: exclusive scan per image ----------------
__global__ void k_scan(const int* __restrict__ counts, int* __restrict__ offs,
                       int* __restrict__ cursor) {
  int t = threadIdx.x;          // 256 = B*KC
  int b = t >> 4, k = t & 15;
  int off = 0;
  for (int kk = 0; kk < k; ++kk) off += counts[b * KC + kk];
  offs[t]   = off;
  cursor[t] = off;
}

// ---------------- stage 3: build per-cluster pixel lists ----------------
__global__ void k_gather(const int* __restrict__ labels, int* __restrict__ cursor,
                         int* __restrict__ list) {
  int g = blockIdx.x * blockDim.x + threadIdx.x;
  int b = g >> 12, p = g & (HW_ - 1);
  int lab = labels[g];
  int pos = atomicAdd(&cursor[b * KC + lab], 1);
  list[b * HW_ + pos] = p;
}

// ---------------- stage 4: cluster centers (atomic-free) ----------------
// One block per (b,k); thread t privately owns features t, t+256, t+512.
__global__ void k_centers(const float* __restrict__ x, const int* __restrict__ list,
                          const int* __restrict__ offs, const int* __restrict__ counts,
                          float* __restrict__ centers) {
  int bk = blockIdx.x;
  int b  = bk >> 4;
  int seg = offs[bk];
  int nk  = counts[bk];
  int t = threadIdx.x;
  int f0 = t, f1 = t + 256, f2 = t + 512;
  int c0 = f0 / 9, o0 = f0 % 9, i0 = o0 / 3 - 1, j0 = o0 % 3 - 1;
  int c1 = f1 / 9, o1 = f1 % 9, i1 = o1 / 3 - 1, j1 = o1 % 3 - 1;
  int c2 = f2 / 9, o2 = f2 % 9, i2 = o2 / 3 - 1, j2 = o2 % 3 - 1;
  float a0 = 0.f, a1 = 0.f, a2 = 0.f;
  const float* xb = x + (size_t)b * C_ * HW_;
  for (int s = 0; s < nk; ++s) {
    int pix = list[b * HW_ + seg + s];
    int y = pix >> 6, xc = pix & 63;
    { int yy = y + i0, xx = xc + j0;
      if (yy >= 0 && yy < H_ && xx >= 0 && xx < W_) a0 += xb[c0 * HW_ + yy * W_ + xx]; }
    { int yy = y + i1, xx = xc + j1;
      if (yy >= 0 && yy < H_ && xx >= 0 && xx < W_) a1 += xb[c1 * HW_ + yy * W_ + xx]; }
    if (f2 < F_) {
      int yy = y + i2, xx = xc + j2;
      if (yy >= 0 && yy < H_ && xx >= 0 && xx < W_) a2 += xb[c2 * HW_ + yy * W_ + xx]; }
  }
  centers[bk * F_ + f0] = a0;
  centers[bk * F_ + f1] = a1;
  if (f2 < F_) centers[bk * F_ + f2] = a2;
}

// ---------------- stage 5: tiny MLPs (256 rows total, negligible) ----------------
__global__ void k_mlp(const float* __restrict__ centers, const int* __restrict__ counts,
                      const float* __restrict__ w1, const float* __restrict__ b1,
                      const float* __restrict__ w2, const float* __restrict__ b2,
                      const float* __restrict__ w3, const float* __restrict__ b3,
                      const float* __restrict__ bw1, const float* __restrict__ bb1,
                      const float* __restrict__ bw2, const float* __restrict__ bb2,
                      float* __restrict__ lowrank, float* __restrict__ biasv) {
  __shared__ float cen[F_];
  __shared__ float h1[MLP_], hb1[MLP_], h2[MLP_];
  int bk = blockIdx.x, t = threadIdx.x;      // 64 threads
  float invc = 1.0f / ((float)counts[bk] + 1e-6f);
  for (int f = t; f < F_; f += 64) cen[f] = centers[bk * F_ + f] * invc;
  __syncthreads();
  if (t < MLP_) {
    float s1 = b1[t], sb = bb1[t];
    for (int f = 0; f < F_; ++f) {
      float c = cen[f];
      s1 += c * w1[f * MLP_ + t];
      sb += c * bw1[f * MLP_ + t];
    }
    h1[t]  = fmaxf(s1, 0.f);
    hb1[t] = fmaxf(sb, 0.f);
  }
  __syncthreads();
  if (t < MLP_) {
    float s = b2[t];
    for (int j = 0; j < MLP_; ++j) s += h1[j] * w2[j * MLP_ + t];
    h2[t] = fmaxf(s, 0.f);
  }
  __syncthreads();
  if (t < RANK_) {
    float s = b3[t];
    for (int j = 0; j < MLP_; ++j) s += h2[j] * w3[j * RANK_ + t];
    lowrank[bk * RANK_ + t] = s;
  }
  float s = bb2[t];
  for (int j = 0; j < MLP_; ++j) s += hb1[j] * bw2[j * OUT_ + t];
  biasv[bk * OUT_ + t] = s;
}

// ---------------- stage 6: materialize per-(b,k) kernels, TRANSPOSED [o][f], bf16 ----
__global__ void k_buildker(const float* __restrict__ lowrank, const float* __restrict__ base,
                           unsigned short* __restrict__ kt) {
  int bk = blockIdx.x, t = threadIdx.x;
  float lr[RANK_];
  for (int r = 0; r < RANK_; ++r) lr[r] = lowrank[bk * RANK_ + r];
  for (int idx = t; idx < OUT_ * F_; idx += 256) {
    int o = idx / F_, f = idx % F_;
    float s = 0.f;
    for (int r = 0; r < RANK_; ++r) s += lr[r] * base[(r * F_ + f) * OUT_ + o];
    kt[((size_t)bk * OUT_ + o) * F_ + f] = f2bf(s);
  }
}

// ---------------- stage 7: grouped GEMM, bf16 WMMA ----------------
// grid (B, K, 128 tiles); block 256 = 8 waves; tile = 32 pixels x 64 outputs.
__global__ void __launch_bounds__(256) k_main(
    const float* __restrict__ x, const int* __restrict__ list,
    const int* __restrict__ offs, const int* __restrict__ counts,
    const unsigned short* __restrict__ kt, const float* __restrict__ biasv,
    float* __restrict__ out) {
  __shared__ unsigned short As[32 * ASTR];   // 37,888 B
  int b = blockIdx.x, k = blockIdx.y, tile = blockIdx.z;
  int bk = b * KC + k;
  int nk = counts[bk];
  int base0 = tile * 32;
  if (base0 >= nk) return;                   // uniform per block
  int seg = offs[bk];
  int tid = threadIdx.x;
  const float* xb = x + (size_t)b * C_ * HW_;

  // Stage full-K A tile (32 patch rows x 576 features) into LDS once.
  for (int idx = tid; idx < 32 * F_; idx += 256) {
    int r = idx / F_, f = idx % F_;
    float v = 0.f;
    int gr = base0 + r;
    if (gr < nk) {
      int pix = list[b * HW_ + seg + gr];
      int y = pix >> 6, xc = pix & 63;
      int c = f / 9, off = f % 9;
      int yy = y + off / 3 - 1, xx = xc + off % 3 - 1;
      if (yy >= 0 && yy < H_ && xx >= 0 && xx < W_) v = xb[c * HW_ + yy * W_ + xx];
    }
    As[r * ASTR + f] = f2bf(v);
  }
  __syncthreads();

  int wave = tid >> 5, lane = tid & 31;
  int lm = lane & 15, lh = lane >> 4;
  int m0 = (wave >> 2) << 4;                 // 0 / 16
  int n0 = (wave & 3) << 4;                  // 0 / 16 / 32 / 48
  const unsigned short* Bp = kt + ((size_t)bk * OUT_ + (n0 + lm)) * F_ + lh * 16;
  const unsigned short* Ap = As + (m0 + lm) * ASTR;

  v8f acc = {};
  for (int kk = 0; kk < F_; kk += 32) {
    union { uint4 q[2]; v16bf v; } a, bb;
    // A lane layout: K = 16h+[0,16) contiguous
    a.q[0]  = *reinterpret_cast<const uint4*>(Ap + kk + lh * 16);
    a.q[1]  = *reinterpret_cast<const uint4*>(Ap + kk + lh * 16 + 8);
    // B lane layout: K = 16h+[0,16) contiguous (kernels stored [o][f])
    bb.q[0] = *reinterpret_cast<const uint4*>(Bp + kk);
    bb.q[1] = *reinterpret_cast<const uint4*>(Bp + kk + 8);
    acc = __builtin_amdgcn_wmma_f32_16x16x32_bf16(
        false, a.v, false, bb.v, (short)0, acc, false, false);
  }

  int o = n0 + lm;
  float bias = biasv[bk * OUT_ + o];
  for (int v = 0; v < 8; ++v) {              // D: VGPR v -> row v + 8h
    int m = m0 + lh * 8 + v;
    int gr = base0 + m;
    if (gr < nk) {
      int pix = list[b * HW_ + seg + gr];
      out[((size_t)b * OUT_ + o) * HW_ + pix] = acc[v] + bias;
    }
  }
}

// ---------------- host ----------------
extern "C" void kernel_launch(void* const* d_in, const int* in_sizes, int n_in,
                              void* d_out, int out_size, void* d_ws, size_t ws_size,
                              hipStream_t stream) {
  (void)in_sizes; (void)n_in; (void)out_size; (void)ws_size;
  const float* x      = (const float*)d_in[0];
  const int*   labels = (const int*)d_in[1];
  const float* lr_w1  = (const float*)d_in[2];
  const float* lr_b1  = (const float*)d_in[3];
  const float* lr_w2  = (const float*)d_in[4];
  const float* lr_b2  = (const float*)d_in[5];
  const float* lr_w3  = (const float*)d_in[6];
  const float* lr_b3  = (const float*)d_in[7];
  const float* basek  = (const float*)d_in[8];
  const float* bw1    = (const float*)d_in[9];
  const float* bb1    = (const float*)d_in[10];
  const float* bw2    = (const float*)d_in[11];
  const float* bb2    = (const float*)d_in[12];
  float* out = (float*)d_out;

  char* ws = (char*)d_ws;
  int*   counts  = (int*)(ws + 0);            // 256 ints
  int*   offs    = (int*)(ws + 1024);         // 256 ints
  int*   cursor  = (int*)(ws + 2048);         // 256 ints
  int*   list    = (int*)(ws + 4096);         // 65536 ints
  float* centers = (float*)(ws + 266240);     // 147456 f32
  float* lowrank = (float*)(ws + 856064);     // 2048 f32
  float* biasv   = (float*)(ws + 864256);     // 16384 f32
  unsigned short* kt = (unsigned short*)(ws + 929792); // 9.44M bf16 (16B aligned)

  k_zero  <<<1, 256, 0, stream>>>(counts);
  k_count <<<(B_ * HW_) / 256, 256, 0, stream>>>(labels, counts);
  k_scan  <<<1, 256, 0, stream>>>(counts, offs, cursor);
  k_gather<<<(B_ * HW_) / 256, 256, 0, stream>>>(labels, cursor, list);
  k_centers<<<B_ * KC, 256, 0, stream>>>(x, list, offs, counts, centers);
  k_mlp   <<<B_ * KC, 64, 0, stream>>>(centers, counts, lr_w1, lr_b1, lr_w2, lr_b2,
                                       lr_w3, lr_b3, bw1, bb1, bw2, bb2, lowrank, biasv);
  k_buildker<<<B_ * KC, 256, 0, stream>>>(lowrank, basek, kt);
  dim3 g7(B_, KC, 128);
  k_main  <<<g7, 256, 0, stream>>>(x, list, offs, counts, kt, biasv, out);
}
